// SepFlexConv_11493332484128
// MI455X (gfx1250) — matrix operand order repaired and checked by reading, compiled, not run on hardware
//
#include <hip/hip_runtime.h>

// ---- problem constants (match reference) ----
#define Bsz    32
#define Cch    128
#define Lseq   4096
#define Hch    128
#define KHID   32
#define NLAY   3
#define OMEGA0 976.0f
#define PADF   128                // front zero pad (covers pipelined windows)
#define LPAD   (Lseq + 192)       // [128, L+127] valid, zeros elsewhere

typedef _Float16 v16h __attribute__((ext_vector_type(16)));
typedef _Float16 v8h  __attribute__((ext_vector_type(8)));
typedef float    v8f  __attribute__((ext_vector_type(8)));

static __device__ __forceinline__ v8f wmma_f16(v16h a, v16h b, v8f c) {
  return __builtin_amdgcn_wmma_f32_16x16x32_f16(false, a, false, b,
                                                (short)0, c, false, false);
}
static __device__ __forceinline__ v16h cat16(v8h lo, v8h hi) {
  return __builtin_shufflevector(lo, hi, 0, 1, 2, 3, 4, 5, 6, 7,
                                         8, 9, 10, 11, 12, 13, 14, 15);
}
static __device__ __forceinline__ v16h ldB(const _Float16* p, int off) {
  v16h r;
  #pragma unroll
  for (int n = 0; n < 16; ++n) r[n] = p[off - n - (n & 8)];
  return r;
}

// -------------------------------------------------------------------------
// Kernel A: x [B,C,L] f32  ->  xT16 [C][B][L] f16  (both sides coalesced)
// -------------------------------------------------------------------------
__global__ __launch_bounds__(256) void k_transpose(const float* __restrict__ x,
                                                   _Float16* __restrict__ xT) {
  size_t idx = (size_t)blockIdx.x * 256 + threadIdx.x;   // = (c*32+b)*L + s
  int s   = (int)(idx & (Lseq - 1));
  int row = (int)(idx >> 12);          // c*32 + b
  int c = row >> 5, b = row & 31;
  xT[idx] = (_Float16)x[((size_t)(b * Cch + c)) * Lseq + s];
}

// -------------------------------------------------------------------------
// Kernel B: MAGNet MFN -> masked conv kernel, stored reversed+padded:
//   G[c][i] = mask(p)*kernel[c][p] at p = L+127-i for i in [128, L+127],
//   zeros elsewhere, so Toeplitz element T[s,t] = g_c[t-s] = G[c][(t-s)+128];
// causal (t-s<0) and pipeline-overrun windows land in genuine zeros.
// -------------------------------------------------------------------------
__global__ __launch_bounds__(256) void k_gen_kernel(
    const float* __restrict__ gw, const float* __restrict__ gb,
    const float* __restrict__ ggam, const float* __restrict__ gmu,
    const float* __restrict__ lw, const float* __restrict__ lb,
    const float* __restrict__ ow, const float* __restrict__ ob,
    const float* __restrict__ mwid, _Float16* __restrict__ G) {
  int i = blockIdx.x * 256 + threadIdx.x;
  if (i >= LPAD) return;
  if (i < PADF || i > Lseq + PADF - 1) {
    for (int c = 0; c < Cch; ++c) G[(size_t)c * LPAD + i] = (_Float16)0.0f;
    return;
  }
  int p = Lseq + PADF - 1 - i;
  float pos = -1.0f + 2.0f * (float)p / (float)(Lseq - 1);

  float h[KHID];
  #pragma unroll 4
  for (int j = 0; j < KHID; ++j) {
    float t = ggam[j] * (pos - gmu[j]);
    h[j] = expf(-0.5f * t * t) * sinf(OMEGA0 * pos * gw[j] + gb[j]);
  }
  for (int layer = 0; layer < NLAY; ++layer) {
    const float* lwr = lw + (size_t)layer * KHID * KHID;
    const float* lbr = lb + layer * KHID;
    int go = (layer + 1) * KHID;
    float nh[KHID];
    for (int j = 0; j < KHID; ++j) {
      float acc = lbr[j];
      #pragma unroll 8
      for (int k = 0; k < KHID; ++k) acc += h[k] * lwr[j * KHID + k];
      float t = ggam[go + j] * (pos - gmu[go + j]);
      nh[j] = expf(-0.5f * t * t) *
              sinf(OMEGA0 * pos * gw[go + j] + gb[go + j]) * acc;
    }
    #pragma unroll 8
    for (int j = 0; j < KHID; ++j) h[j] = nh[j];
  }
  float mw = mwid[0];
  float dm = pos - 1.0f;                                  // mask_mean = 1.0
  float maskv = expf(-0.5f * dm * dm / (mw * mw + 1e-8f));
  float norm = rsqrtf((float)(Cch * Lseq));
  for (int c = 0; c < Cch; ++c) {
    float acc = 0.0f;
    const float* owr = ow + c * KHID;
    #pragma unroll 8
    for (int j = 0; j < KHID; ++j) acc += h[j] * owr[j];
    G[(size_t)c * LPAD + i] = (_Float16)((acc * norm + ob[c]) * maskv);
  }
}

// -------------------------------------------------------------------------
// small f32 -> f16 convert (pointwise weights)
// -------------------------------------------------------------------------
__global__ __launch_bounds__(256) void k_cvt(const float* __restrict__ src,
                                             _Float16* __restrict__ dst, int n) {
  int i = blockIdx.x * 256 + threadIdx.x;
  if (i < n) dst[i] = (_Float16)src[i];
}

// -------------------------------------------------------------------------
// Kernel C: causal depthwise conv as Toeplitz WMMA matmul.
// Wave: 4 accumulators = 4 adjacent t-tiles (64 t) x one 16-batch tile.
// One A load (x, 16b x 32s) feeds 4 WMMAs; the 4 Toeplitz B tiles are 64
// consecutive halves of the LDS-staged reversed kernel row. Both the A
// (global) and B (LDS) operands are software-pipelined one k-step ahead,
// unconditionally (overruns land in allocated memory / zero pad).
// Block = 8 waves: (4 x 64-t groups) x (2 b-tiles), one channel, 256 t.
// -------------------------------------------------------------------------
__global__ __launch_bounds__(256) void k_toeplitz_wmma(
    const _Float16* __restrict__ xT, const _Float16* __restrict__ G,
    const float* __restrict__ cbias, _Float16* __restrict__ y16) {
  __shared__ _Float16 Gs[LPAD];                       // 8576 B
  int c = blockIdx.y;
  for (int i = threadIdx.x; i < LPAD; i += 256)
    Gs[i] = G[(size_t)c * LPAD + i];
  __syncthreads();

  int w     = threadIdx.x >> 5;                       // wave in block
  int l     = threadIdx.x & 31;                       // lane
  int b0    = (w & 1) * 16;
  int tbase = blockIdx.x * 256 + (w >> 1) * 64;       // 4 t-tiles per wave
  int m     = l & 15;                                 // A row (batch in tile)
  int g     = l >> 4;                                 // A K-chunk selector
  int kB    = 8 * g - m;                              // B Toeplitz offset (k base - column)

  const _Float16* xr = xT + ((size_t)(c * 32 + b0 + m)) * Lseq + 8 * g;
  const _Float16* gq = Gs + (tbase - kB + PADF);      // Toeplitz window base

  v8f acc0 = {}, acc1 = {}, acc2 = {}, acc3 = {};
  int nk = __builtin_amdgcn_readfirstlane(((tbase + 48) >> 5) + 1);

  v8h lo = *(const v8h*)(xr);
  v8h hi = *(const v8h*)(xr + 16);
  v16h bm0 = ldB(gq, 0),  bm1 = ldB(gq, 16);
  v16h bm2 = ldB(gq, 32), bm3 = ldB(gq, 48);

  for (int kk = 0; kk < nk; ++kk) {
    v16h a = cat16(lo, hi);
    int s1 = (kk + 1) << 5;
    lo = *(const v8h*)(xr + s1);                      // next A (pipelined)
    hi = *(const v8h*)(xr + s1 + 16);
    __builtin_prefetch(xr + s1 + 32, 0, 3);           // global_prefetch_b8
    acc0 = wmma_f16(a, bm0, acc0);
    acc1 = wmma_f16(a, bm1, acc1);
    acc2 = wmma_f16(a, bm2, acc2);
    acc3 = wmma_f16(a, bm3, acc3);
    gq -= 32;                                         // next Toeplitz window
    bm0 = ldB(gq, 0);  bm1 = ldB(gq, 16);             // next B (pipelined)
    bm2 = ldB(gq, 32); bm3 = ldB(gq, 48);
  }

  float cb = cbias[c];
  int n = l & 15, hiHalf = l >> 4;
  #pragma unroll
  for (int v = 0; v < 8; ++v) {
    size_t row = ((size_t)(c * 32 + b0 + v + 8 * hiHalf)) * Lseq + n;
    y16[row + tbase     ] = (_Float16)(acc0[v] + cb);
    y16[row + tbase + 16] = (_Float16)(acc1[v] + cb);
    y16[row + tbase + 32] = (_Float16)(acc2[v] + cb);
    y16[row + tbase + 48] = (_Float16)(acc3[v] + cb);
  }
}

// -------------------------------------------------------------------------
// Kernel D: pointwise conv  out[b,h,t] = sum_c pw[h,c]*y[b,c,t] + pw_b[h]
// as WMMA GEMM: A = pw16 (16h x 32c), B = y16 (32c x 16t at fixed b).
// Block = 8 waves = 8 h-tiles; grid = (t-tiles, batch).
// -------------------------------------------------------------------------
__global__ __launch_bounds__(256) void k_pointwise_wmma(
    const _Float16* __restrict__ pw16, const _Float16* __restrict__ y16,
    const float* __restrict__ pwb, float* __restrict__ out) {
  int b  = blockIdx.y;
  int t0 = blockIdx.x * 16;
  int w  = threadIdx.x >> 5;
  int l  = threadIdx.x & 31;
  int h0 = w * 16;
  int m  = l & 15;
  int g  = l >> 4;
  int kB = m + 16 * g;

  const _Float16* pwr = pw16 + (size_t)(h0 + m) * Cch;

  v8f acc = {};
  #pragma unroll
  for (int kk = 0; kk < Cch / 32; ++kk) {
    int c0 = kk << 5;
    v8h lo = *(const v8h*)(pwr + c0 + 8 * g);
    v8h hi = *(const v8h*)(pwr + c0 + 16 + 8 * g);
    v16h a = cat16(lo, hi);
    v16h bm;
    for (int e = 0; e < 16; ++e) bm[e] = y16[((size_t)((c0 + 8 * g + e + (e & 8)) * 32 + b)) * Lseq + t0 + m];
    acc = wmma_f16(a, bm, acc);
  }

  int n = l & 15, hiHalf = l >> 4;
  #pragma unroll
  for (int v = 0; v < 8; ++v) {
    int hh = h0 + v + 8 * hiHalf;
    out[((size_t)(b * Hch + hh)) * Lseq + t0 + n] = acc[v] + pwb[hh];
  }
}

// -------------------------------------------------------------------------
extern "C" void kernel_launch(void* const* d_in, const int* in_sizes, int n_in,
                              void* d_out, int out_size, void* d_ws, size_t ws_size,
                              hipStream_t stream) {
  (void)in_sizes; (void)n_in; (void)out_size; (void)ws_size;
  const float* x     = (const float*)d_in[0];
  const float* g_w   = (const float*)d_in[1];
  const float* g_b   = (const float*)d_in[2];
  const float* g_gam = (const float*)d_in[3];
  const float* g_mu  = (const float*)d_in[4];
  const float* lin_w = (const float*)d_in[5];
  const float* lin_b = (const float*)d_in[6];
  const float* out_w = (const float*)d_in[7];
  const float* out_b = (const float*)d_in[8];
  const float* mwid  = (const float*)d_in[9];
  const float* cbias = (const float*)d_in[10];
  const float* pw_w  = (const float*)d_in[11];
  const float* pw_b  = (const float*)d_in[12];

  char* ws = (char*)d_ws;
  const size_t xT_bytes = (size_t)Cch * Bsz * Lseq * sizeof(_Float16);  // 32 MB
  const size_t G_bytes  = (size_t)Cch * LPAD * sizeof(_Float16);        // ~1 MB
  const size_t pw_bytes = (size_t)Hch * Cch * sizeof(_Float16);         // 32 KB
  _Float16* xT16 = (_Float16*)(ws);
  _Float16* G    = (_Float16*)(ws + xT_bytes);
  _Float16* pw16 = (_Float16*)(ws + xT_bytes + G_bytes);
  _Float16* y16  = (_Float16*)(ws + xT_bytes + G_bytes + pw_bytes);     // 32 MB

  // 1) transpose/convert x -> [C][B][L] f16
  k_transpose<<<(Cch * Bsz * Lseq) / 256, 256, 0, stream>>>(x, xT16);
  // 2) MFN kernel generation (reversed + padded, mask applied)
  k_gen_kernel<<<(LPAD + 255) / 256, 256, 0, stream>>>(
      g_w, g_b, g_gam, g_mu, lin_w, lin_b, out_w, out_b, mwid, G);
  // 3) pw weights -> f16
  k_cvt<<<(Hch * Cch + 255) / 256, 256, 0, stream>>>(pw_w, pw16, Hch * Cch);
  // 4) causal depthwise conv via Toeplitz WMMA (4 t-tiles per wave)
  dim3 gc(Lseq / 256, Cch);
  k_toeplitz_wmma<<<gc, 256, 0, stream>>>(xT16, G, cbias, y16);
  // 5) pointwise conv via WMMA GEMM
  dim3 gd(Lseq / 16, Bsz);
  k_pointwise_wmma<<<gd, 256, 0, stream>>>(pw16, y16, pw_b, (float*)d_out);
}
